// LRULayer_22368189677853
// MI455X (gfx1250) — matrix operand order repaired and checked by reading, not compiled, not run on hardware
//
#include <hip/hip_runtime.h>

// ---------------------------------------------------------------------------
// LRU layer for MI455X (gfx1250, wave32, WMMA).
//   STATE_N = HID_H = 256, BATCH = 16, SEQ_L = 4096.
// Pipeline: prep -> Bu GEMM (bf16 WMMA) -> chunk carries -> carry scan ->
//           fused chunk apply + output GEMM (bf16 WMMA) + D*u skip.
// Round 2: process two 16x16 output tiles per iteration in both GEMMs so the
// WMMA stream has more independent accumulator chains (4 in K1, 2 in K3) (hides XDL + L2 latency).
// ---------------------------------------------------------------------------

#define STATE_N 256
#define HID_H   256
#define BATCH_B 16
#define SEQ_L   4096
#define MTOT    (BATCH_B * SEQ_L)   // 65536 rows total
#define CHUNK   64
#define NCHUNK  (SEQ_L / CHUNK)     // 64 chunks per sequence
#define LDS_PITCH 264               // 256 + 8 ushort pad (kills LDS bank conflicts)

typedef __attribute__((ext_vector_type(16))) __bf16        v16bf;
typedef __attribute__((ext_vector_type(8)))  float         v8f;
typedef __attribute__((ext_vector_type(4)))  unsigned int  u32x4;
typedef __attribute__((ext_vector_type(2)))  unsigned int  u32x2;
typedef __attribute__((ext_vector_type(4)))  float         f32x4;

union FragU { u32x4 u[2]; v16bf v; };   // 8 dwords == one 16-lane bf16 WMMA operand

__device__ __forceinline__ unsigned int f2bf(float f) {   // fp32 -> bf16 (RNE)
  unsigned int u = __float_as_uint(f);
  u += 0x7FFFu + ((u >> 16) & 1u);
  return u >> 16;
}
__device__ __forceinline__ unsigned int pack2(unsigned int lo, unsigned int hi) {
  return (lo & 0xFFFFu) | (hi << 16);
}

// ---------------------------------------------------------------------------
// K0: build Lambda, Lambda^64, and pre-swizzled bf16 WMMA B-operand fragments
//     for B_norm (Bu GEMM) and C / -C_im (output GEMM).
// Fragment layout: frag[ks][nt][lane][j] (j fastest) => each lane reads its
// 8 dwords with two 16B loads; fully coalesced; L2-resident (128KB each).
// Assumed dense 16-bit B-operand (32x16) layout: lanes 0-15 hold K=0..7,16..23
// (Vj packs K=2j,2j+1; +8 for j>=4), lanes 16-31 hold K=8..15,24..31; column = lane&15.
// ---------------------------------------------------------------------------
__global__ void lru_prep(const float* __restrict__ nu_log, const float* __restrict__ theta_log,
                         const float* __restrict__ B_re, const float* __restrict__ B_im,
                         const float* __restrict__ C_re, const float* __restrict__ C_im,
                         const float* __restrict__ gamma_log,
                         float* __restrict__ lam_re, float* __restrict__ lam_im,
                         float* __restrict__ lam64_re, float* __restrict__ lam64_im,
                         unsigned int* __restrict__ bnfrag_re, unsigned int* __restrict__ bnfrag_im,
                         unsigned int* __restrict__ ctfrag_re, unsigned int* __restrict__ ctfrag_imn) {
  __shared__ float gam_s[STATE_N];
  const int tid = threadIdx.x;
  {
    const int n = tid;
    float nu  = __expf(nu_log[n]);
    float th  = __expf(theta_log[n]);
    float mag = __expf(-nu);
    float lr  = mag * __cosf(th);
    float li  = mag * __sinf(th);
    lam_re[n] = lr; lam_im[n] = li;
    float ar = lr, ai = li;                 // Lambda^64 by 6 complex squarings
#pragma unroll
    for (int i = 0; i < 6; i++) { float nr = ar*ar - ai*ai; float ni = 2.f*ar*ai; ar = nr; ai = ni; }
    lam64_re[n] = ar; lam64_im[n] = ai;
    gam_s[n] = __expf(gamma_log[n]);
  }
  __syncthreads();
  for (int f = tid; f < 8 * 16 * 32 * 8; f += 256) {
    const int j    = f & 7;
    const int lane = (f >> 3) & 31;
    const int nt   = (f >> 8) & 15;
    const int ks   = f >> 12;
    const int col  = nt * 16 + (lane & 15);                    // n for B_norm, h for C
    const int k    = ks * 32 + ((lane < 16) ? 0 : 8) + 2 * j + ((j < 4) ? 0 : 8); // reduction index
    // Bu GEMM B-operand: element (k=h, col=n) = B_norm[n][h] = (B_re+iB_im)[n][h]*gamma[n]
    const float g = gam_s[col];
    bnfrag_re[f] = pack2(f2bf(B_re[col * HID_H + k] * g), f2bf(B_re[col * HID_H + k + 1] * g));
    bnfrag_im[f] = pack2(f2bf(B_im[col * HID_H + k] * g), f2bf(B_im[col * HID_H + k + 1] * g));
    // Output GEMM B-operand: element (k=n, col=h) = C_re[h][n]  and  -C_im[h][n]
    ctfrag_re[f]  = pack2(f2bf( C_re[col * STATE_N + k]), f2bf( C_re[col * STATE_N + k + 1]));
    ctfrag_imn[f] = pack2(f2bf(-C_im[col * STATE_N + k]), f2bf(-C_im[col * STATE_N + k + 1]));
  }
}

// ---------------------------------------------------------------------------
// K1: Bu = x @ B_norm^T  (real x complex => two bf16 WMMA GEMMs sharing A).
// 512 workgroups x 128 rows; A tile staged fp32->bf16 in padded LDS.
// A-fragment per 16-bit ISA layout: lanes<16 read K {0..7,16..23},
// lanes>=16 read K {8..15,24..31} of their row -> two ds b128 loads.
// Two n-tiles per iteration => 4 independent WMMA accumulator chains.
// ---------------------------------------------------------------------------
__global__ void __launch_bounds__(256)
lru_bu_gemm(const float* __restrict__ x,
            const unsigned int* __restrict__ bnfrag_re,
            const unsigned int* __restrict__ bnfrag_im,
            float* __restrict__ bu_re, float* __restrict__ bu_im) {
  extern __shared__ unsigned short ldsA[];            // [128][LDS_PITCH]
  const int tid = threadIdx.x;
  const int m0g = blockIdx.x * 128;
  {                                                   // stage + convert A tile
    const int r  = tid >> 1;
    const int c0 = (tid & 1) * 128;
    const f32x4* src = (const f32x4*)(x + (size_t)(m0g + r) * HID_H + c0);
    unsigned short* dstrow = &ldsA[r * LDS_PITCH + c0];
#pragma unroll 4
    for (int i = 0; i < 32; i++) {
      f32x4 v = src[i];
      u32x2 p;
      p.x = pack2(f2bf(v.x), f2bf(v.y));
      p.y = pack2(f2bf(v.z), f2bf(v.w));
      *(u32x2*)&dstrow[i * 4] = p;
    }
  }
  __syncthreads();
  const int w = tid >> 5, lane = tid & 31;
  const int m0l = w * 16;
  const int hiHalf = (lane < 16) ? 0 : 8;
  const unsigned short* arow = &ldsA[(m0l + (lane & 15)) * LDS_PITCH];

  FragU afr[8];                                       // hoist all 8 A k-fragments
#pragma unroll
  for (int ks = 0; ks < 8; ks++) {
    const int kb = ks * 32 + hiHalf;
    afr[ks].u[0] = *(const u32x4*)&arow[kb];
    afr[ks].u[1] = *(const u32x4*)&arow[kb + 16];
  }

  for (int nt = 0; nt < 16; nt += 2) {
    v8f acc_re0 = {}; v8f acc_im0 = {};
    v8f acc_re1 = {}; v8f acc_im1 = {};
#pragma unroll
    for (int ks = 0; ks < 8; ks++) {
      FragU br0, bi0, br1, bi1;
      const size_t fb = ((size_t)(ks * 16 + nt) * 32 + lane) * 2;
      const u32x4* pr0 = (const u32x4*)bnfrag_re + fb;
      const u32x4* pi0 = (const u32x4*)bnfrag_im + fb;
      br0.u[0] = pr0[0];  br0.u[1] = pr0[1];
      bi0.u[0] = pi0[0];  bi0.u[1] = pi0[1];
      br1.u[0] = pr0[64]; br1.u[1] = pr0[65];        // nt+1 => +32 lanes *2 uint4
      bi1.u[0] = pi0[64]; bi1.u[1] = pi0[65];
      acc_re0 = __builtin_amdgcn_wmma_f32_16x16x32_bf16(false, afr[ks].v, false, br0.v, (short)0, acc_re0, false, false);
      acc_im0 = __builtin_amdgcn_wmma_f32_16x16x32_bf16(false, afr[ks].v, false, bi0.v, (short)0, acc_im0, false, false);
      acc_re1 = __builtin_amdgcn_wmma_f32_16x16x32_bf16(false, afr[ks].v, false, br1.v, (short)0, acc_re1, false, false);
      acc_im1 = __builtin_amdgcn_wmma_f32_16x16x32_bf16(false, afr[ks].v, false, bi1.v, (short)0, acc_im1, false, false);
    }
    const int col = nt * 16 + (lane & 15);
    const int rb  = m0g + m0l + hiHalf;                // D-layout: VGPR i -> row i / 8+i
#pragma unroll
    for (int i = 0; i < 8; i++) {
      const size_t off = (size_t)(rb + i) * STATE_N + col;
      bu_re[off]      = acc_re0[i];
      bu_im[off]      = acc_im0[i];
      bu_re[off + 16] = acc_re1[i];
      bu_im[off + 16] = acc_im1[i];
    }
  }
}

// ---------------------------------------------------------------------------
// K2a: per-(batch, chunk, channel) local complex scan carry (zero init).
// ---------------------------------------------------------------------------
__global__ void lru_chunk_carry(const float* __restrict__ bu_re, const float* __restrict__ bu_im,
                                const float* __restrict__ lam_re, const float* __restrict__ lam_im,
                                float* __restrict__ carry_re, float* __restrict__ carry_im) {
  const int gid   = blockIdx.x * 256 + threadIdx.x;
  const int chan  = gid & 255;
  const int chunk = (gid >> 8) & 63;
  const int b     = gid >> 14;
  const float lr = lam_re[chan], li = lam_im[chan];
  float s_r = 0.f, s_i = 0.f;
  const size_t base = ((size_t)(b * SEQ_L + chunk * CHUNK)) * STATE_N + chan;
  for (int t = 0; t < CHUNK; t++) {
    const float br = bu_re[base + (size_t)t * STATE_N];
    const float bi = bu_im[base + (size_t)t * STATE_N];
    const float nr = fmaf(lr, s_r, fmaf(-li, s_i, br));
    const float ni = fmaf(lr, s_i, fmaf( li, s_r, bi));
    s_r = nr; s_i = ni;
  }
  const size_t cidx = (size_t)(b * NCHUNK + chunk) * STATE_N + chan;
  carry_re[cidx] = s_r; carry_im[cidx] = s_i;
}

// ---------------------------------------------------------------------------
// K2b: exclusive scan of chunk carries with Lambda^64 -> per-chunk prefix.
// ---------------------------------------------------------------------------
__global__ void lru_carry_scan(const float* __restrict__ carry_re, const float* __restrict__ carry_im,
                               const float* __restrict__ lam64_re, const float* __restrict__ lam64_im,
                               float* __restrict__ prefix_re, float* __restrict__ prefix_im) {
  const int b = blockIdx.x, chan = threadIdx.x;
  const float ar = lam64_re[chan], ai = lam64_im[chan];
  float p_r = 0.f, p_i = 0.f;
  for (int c = 0; c < NCHUNK; c++) {
    const size_t idx = (size_t)(b * NCHUNK + c) * STATE_N + chan;
    prefix_re[idx] = p_r; prefix_im[idx] = p_i;
    const float cr = carry_re[idx], ci = carry_im[idx];
    const float nr = fmaf(ar, p_r, fmaf(-ai, p_i, cr));
    const float ni = fmaf(ar, p_i, fmaf( ai, p_r, ci));
    p_r = nr; p_i = ni;
  }
}

// ---------------------------------------------------------------------------
// K3: fused chunk apply + output GEMM. Phase A: re-run the 64-step scan
// seeded with the chunk prefix, deposit bf16 states in LDS. Phase B:
// out = Re(S @ C^T) + D*u via two WMMAs/k-step (C_im pre-negated).
// Two h-tiles per iteration => 2 independent WMMA accumulator chains.
// ---------------------------------------------------------------------------
__global__ void __launch_bounds__(256)
lru_apply_out(const float* __restrict__ x,
              const float* __restrict__ bu_re, const float* __restrict__ bu_im,
              const float* __restrict__ prefix_re, const float* __restrict__ prefix_im,
              const float* __restrict__ lam_re, const float* __restrict__ lam_im,
              const unsigned int* __restrict__ ctfrag_re,
              const unsigned int* __restrict__ ctfrag_imn,
              const float* __restrict__ Dv,
              float* __restrict__ out) {
  extern __shared__ unsigned short ldsS[];            // s_re[64][264] then s_im[64][264]
  unsigned short* sre = ldsS;
  unsigned short* sim = ldsS + CHUNK * LDS_PITCH;
  const int tid   = threadIdx.x;
  const int b     = blockIdx.x >> 6;
  const int chunk = blockIdx.x & 63;
  {                                                   // phase A: one channel per thread
    const int chan = tid;
    const float lr = lam_re[chan], li = lam_im[chan];
    const size_t pidx = (size_t)(b * NCHUNK + chunk) * STATE_N + chan;
    float s_r = prefix_re[pidx], s_i = prefix_im[pidx];
    const size_t base = ((size_t)(b * SEQ_L + chunk * CHUNK)) * STATE_N + chan;
    for (int t = 0; t < CHUNK; t++) {
      const float br = bu_re[base + (size_t)t * STATE_N];
      const float bi = bu_im[base + (size_t)t * STATE_N];
      const float nr = fmaf(lr, s_r, fmaf(-li, s_i, br));
      const float ni = fmaf(lr, s_i, fmaf( li, s_r, bi));
      s_r = nr; s_i = ni;
      sre[t * LDS_PITCH + chan] = (unsigned short)f2bf(s_r);
      sim[t * LDS_PITCH + chan] = (unsigned short)f2bf(s_i);
    }
  }
  __syncthreads();
  const int w = tid >> 5, lane = tid & 31;
  const int m0l = (w & 3) * 16;                        // 4 row-subtiles x 2 h-halves
  const int htb = (w >> 2) * 8;
  const int hiHalf = (lane < 16) ? 0 : 8;
  const unsigned short* arow_re = &sre[(m0l + (lane & 15)) * LDS_PITCH];
  const unsigned short* arow_im = &sim[(m0l + (lane & 15)) * LDS_PITCH];

  FragU afr_re[8], afr_im[8];                          // hoist state k-fragments
#pragma unroll
  for (int ks = 0; ks < 8; ks++) {
    const int kb = ks * 32 + hiHalf;
    afr_re[ks].u[0] = *(const u32x4*)&arow_re[kb];
    afr_re[ks].u[1] = *(const u32x4*)&arow_re[kb + 16];
    afr_im[ks].u[0] = *(const u32x4*)&arow_im[kb];
    afr_im[ks].u[1] = *(const u32x4*)&arow_im[kb + 16];
  }

  for (int jt = 0; jt < 8; jt += 2) {
    const int ht = htb + jt;
    v8f acc0 = {}; v8f acc1 = {};
#pragma unroll
    for (int ks = 0; ks < 8; ks++) {
      FragU cr0, ci0, cr1, ci1;
      const size_t fb = ((size_t)(ks * 16 + ht) * 32 + lane) * 2;
      const u32x4* pr = (const u32x4*)ctfrag_re  + fb;
      const u32x4* pi = (const u32x4*)ctfrag_imn + fb;
      cr0.u[0] = pr[0];  cr0.u[1] = pr[1];
      ci0.u[0] = pi[0];  ci0.u[1] = pi[1];
      cr1.u[0] = pr[64]; cr1.u[1] = pr[65];          // ht+1 => +32 lanes *2 uint4
      ci1.u[0] = pi[64]; ci1.u[1] = pi[65];
      acc0 = __builtin_amdgcn_wmma_f32_16x16x32_bf16(false, afr_re[ks].v, false, cr0.v, (short)0, acc0, false, false);
      acc0 = __builtin_amdgcn_wmma_f32_16x16x32_bf16(false, afr_im[ks].v, false, ci0.v, (short)0, acc0, false, false);
      acc1 = __builtin_amdgcn_wmma_f32_16x16x32_bf16(false, afr_re[ks].v, false, cr1.v, (short)0, acc1, false, false);
      acc1 = __builtin_amdgcn_wmma_f32_16x16x32_bf16(false, afr_im[ks].v, false, ci1.v, (short)0, acc1, false, false);
    }
    const int h0 = ht * 16 + (lane & 15);
    const float dv0 = Dv[h0];
    const float dv1 = Dv[h0 + 16];
    const size_t mg = (size_t)(b * SEQ_L + chunk * CHUNK + m0l + hiHalf);
#pragma unroll
    for (int i = 0; i < 8; i++) {
      const size_t off = (mg + i) * HID_H + h0;
      out[off]      = acc0[i] + dv0 * x[off];
      out[off + 16] = acc1[i] + dv1 * x[off + 16];
    }
  }
}

// ---------------------------------------------------------------------------
extern "C" void kernel_launch(void* const* d_in, const int* in_sizes, int n_in,
                              void* d_out, int out_size, void* d_ws, size_t ws_size,
                              hipStream_t stream) {
  const float* x         = (const float*)d_in[0];
  const float* nu_log    = (const float*)d_in[1];
  const float* theta_log = (const float*)d_in[2];
  const float* B_re      = (const float*)d_in[3];
  const float* B_im      = (const float*)d_in[4];
  const float* C_re      = (const float*)d_in[5];
  const float* C_im      = (const float*)d_in[6];
  const float* Dv        = (const float*)d_in[7];
  const float* gamma_log = (const float*)d_in[8];
  float* out = (float*)d_out;

  char* ws = (char*)d_ws;
  size_t off = 0;
  auto alloc = [&](size_t bytes) -> char* {
    char* p = ws + off;
    off = (off + bytes + 255) & ~(size_t)255;
    return p;
  };
  float* bu_re      = (float*)alloc((size_t)MTOT * STATE_N * sizeof(float));   // 64 MB
  float* bu_im      = (float*)alloc((size_t)MTOT * STATE_N * sizeof(float));   // 64 MB
  float* carry_re   = (float*)alloc((size_t)BATCH_B * NCHUNK * STATE_N * sizeof(float));
  float* carry_im   = (float*)alloc((size_t)BATCH_B * NCHUNK * STATE_N * sizeof(float));
  float* prefix_re  = (float*)alloc((size_t)BATCH_B * NCHUNK * STATE_N * sizeof(float));
  float* prefix_im  = (float*)alloc((size_t)BATCH_B * NCHUNK * STATE_N * sizeof(float));
  unsigned int* bnfrag_re  = (unsigned int*)alloc(8 * 16 * 32 * 8 * sizeof(unsigned int));
  unsigned int* bnfrag_im  = (unsigned int*)alloc(8 * 16 * 32 * 8 * sizeof(unsigned int));
  unsigned int* ctfrag_re  = (unsigned int*)alloc(8 * 16 * 32 * 8 * sizeof(unsigned int));
  unsigned int* ctfrag_imn = (unsigned int*)alloc(8 * 16 * 32 * 8 * sizeof(unsigned int));
  float* lam_re   = (float*)alloc(STATE_N * sizeof(float));
  float* lam_im   = (float*)alloc(STATE_N * sizeof(float));
  float* lam64_re = (float*)alloc(STATE_N * sizeof(float));
  float* lam64_im = (float*)alloc(STATE_N * sizeof(float));

  const size_t ldsK1 = (size_t)128 * LDS_PITCH * sizeof(unsigned short);           // 67584 B
  const size_t ldsK3 = (size_t)2 * CHUNK * LDS_PITCH * sizeof(unsigned short);     // 67584 B

  lru_prep<<<1, 256, 0, stream>>>(nu_log, theta_log, B_re, B_im, C_re, C_im, gamma_log,
                                  lam_re, lam_im, lam64_re, lam64_im,
                                  bnfrag_re, bnfrag_im, ctfrag_re, ctfrag_imn);
  lru_bu_gemm<<<MTOT / 128, 256, ldsK1, stream>>>(x, bnfrag_re, bnfrag_im, bu_re, bu_im);
  lru_chunk_carry<<<(BATCH_B * NCHUNK * STATE_N) / 256, 256, 0, stream>>>(
      bu_re, bu_im, lam_re, lam_im, carry_re, carry_im);
  lru_carry_scan<<<BATCH_B, 256, 0, stream>>>(carry_re, carry_im, lam64_re, lam64_im,
                                              prefix_re, prefix_im);
  lru_apply_out<<<BATCH_B * NCHUNK, 256, ldsK3, stream>>>(
      x, bu_re, bu_im, prefix_re, prefix_im, lam_re, lam_im,
      ctfrag_re, ctfrag_imn, Dv, out);
}
